// HybridMambaGPT_23957327577560
// MI455X (gfx1250) — compile-verified, matrix operand order checked by reading, not run on hardware
//
#include <hip/hip_runtime.h>
#include <hip/hip_bf16.h>

// ---------------------------------------------------------------------------
// Types for CDNA5 WMMA (wave32): 16x16x32 bf16 -> f32
// ---------------------------------------------------------------------------
typedef __attribute__((ext_vector_type(16))) __bf16 v16bf;
typedef __attribute__((ext_vector_type(8)))  float  v8f;
typedef __attribute__((ext_vector_type(4)))  unsigned int u32x4;
typedef __attribute__((ext_vector_type(8)))  int   i32x8;
typedef __attribute__((ext_vector_type(4)))  int   i32x4;

__device__ __forceinline__ v8f wmma_bf(v16bf a, v16bf b, v8f c) {
    return __builtin_amdgcn_wmma_f32_16x16x32_bf16(
        false, a, false, b, (short)0, c, false, false);
}

// Use the Tensor Data Mover when the builtin is available.
#if defined(__has_builtin)
#if __has_builtin(__builtin_amdgcn_tensor_load_to_lds)
#define USE_TDM 1
#endif
#endif
#ifndef USE_TDM
#define USE_TDM 0
#endif

// ---------------------------------------------------------------------------
// Model constants
// ---------------------------------------------------------------------------
#define BB_     2
#define SS_     2048
#define DD_     1024
#define TOKENS  (BB_ * SS_)          // 4096
#define D_INNER 2048
#define NHEADS  32
#define HEADDIM 64
#define D_STATE 128
#define NCHUNK  32                   // 2048 / 64
#define CHUNKL  64
#define D_PROJ  4384                 // 2*2048 + 2*128 + 32
#define MLPD    3072
#define AH_     16                   // attention heads
#define AKV_    4                    // kv heads
#define AHD_    64

// zxbcdt row layout offsets
#define OFF_Z   0
#define OFF_XS  2048
#define OFF_B   4096
#define OFF_C   4224
#define OFF_DT  4352

// ---------------------------------------------------------------------------
// Block reduce helper (256 threads, 8 waves of 32)
// ---------------------------------------------------------------------------
__device__ __forceinline__ float block_reduce_sum(float v, float* sbuf) {
    __syncthreads();
#pragma unroll
    for (int off = 16; off; off >>= 1) v += __shfl_xor(v, off, 32);
    int wid = threadIdx.x >> 5, lid = threadIdx.x & 31;
    if (lid == 0) sbuf[wid] = v;
    __syncthreads();
    float r = (threadIdx.x < (blockDim.x >> 5)) ? sbuf[threadIdx.x] : 0.f;
    if (threadIdx.x < 32) {
#pragma unroll
        for (int off = 4; off; off >>= 1) r += __shfl_xor(r, off, 32);
        if (threadIdx.x == 0) sbuf[0] = r;
    }
    __syncthreads();
    return sbuf[0];
}

// ---------------------------------------------------------------------------
// Elementwise conversions
// ---------------------------------------------------------------------------
__global__ void cvt_bf16_k(const float* __restrict__ x, __bf16* __restrict__ o, size_t n) {
    size_t i = (size_t)blockIdx.x * blockDim.x + threadIdx.x;
    if (i < n) o[i] = (__bf16)x[i];
}

__global__ void silu_cvt_k(const float* __restrict__ x, __bf16* __restrict__ o, size_t n) {
    size_t i = (size_t)blockIdx.x * blockDim.x + threadIdx.x;
    if (i < n) { float v = x[i]; o[i] = (__bf16)(v / (1.f + __expf(-v))); }
}

// ---------------------------------------------------------------------------
// RMSNorm / LayerNorm (fp32 in, bf16 out) — one block per token
// ---------------------------------------------------------------------------
__global__ __launch_bounds__(256) void rmsnorm_bf16_k(
    const float* __restrict__ x, const float* __restrict__ w,
    __bf16* __restrict__ out, int D)
{
    __shared__ float sbuf[16];
    int t = blockIdx.x;
    const float* row = x + (size_t)t * D;
    float ss = 0.f;
    for (int i = threadIdx.x; i < D; i += blockDim.x) { float v = row[i]; ss += v * v; }
    ss = block_reduce_sum(ss, sbuf);
    float rstd = rsqrtf(ss / (float)D + 1e-5f);
    for (int i = threadIdx.x; i < D; i += blockDim.x)
        out[(size_t)t * D + i] = (__bf16)(row[i] * rstd * w[i]);
}

__global__ __launch_bounds__(256) void layernorm_bf16_k(
    const float* __restrict__ x, const float* __restrict__ w,
    const float* __restrict__ bia, __bf16* __restrict__ out, int D)
{
    __shared__ float sbuf[16];
    int t = blockIdx.x;
    const float* row = x + (size_t)t * D;
    float s = 0.f, ss = 0.f;
    for (int i = threadIdx.x; i < D; i += blockDim.x) { float v = row[i]; s += v; ss += v * v; }
    s  = block_reduce_sum(s,  sbuf);
    ss = block_reduce_sum(ss, sbuf);
    float mean = s / (float)D;
    float var  = ss / (float)D - mean * mean;
    float rstd = rsqrtf(var + 1e-5f);
    for (int i = threadIdx.x; i < D; i += blockDim.x)
        out[(size_t)t * D + i] = (__bf16)((row[i] - mean) * rstd * w[i] + bia[i]);
}

// ---------------------------------------------------------------------------
// Generic WMMA GEMM:  C[M,N] (f32) = A[M,K] (bf16 row-major) * W[N,K]^T (bf16)
//                     (+ optional fp32 residual add).
// Block = 256 thr (8 waves), tile 128x128, K-step 32.
// TDM version is double-buffered: wave 0 issues the DMA pair for K-step k+1
// into the alternate LDS buffer, then does a *partial* wait
// (s_wait_tensorcnt 2 -> current pair landed, next pair still in flight) so
// the Tensor Data Mover streams tiles while the matrix pipe computes.
// D# pad fields (pad_interval=3 -> every 16 DWORDs, pad_amount=7 -> +8 DWORDs)
// reproduce the 48-element padded LDS row stride the fragments expect.
// ---------------------------------------------------------------------------
#define GPAD 48   // LDS row stride in elems (96B, 32B-aligned fragments)

#if USE_TDM
__device__ __forceinline__ void tdm_load_tile_128x32(
    const __bf16* gsrc, unsigned int lds_addr, int row_stride_elems)
{
    unsigned long long ga = (unsigned long long)(uintptr_t)gsrc;
    // D# group 0: count=1 | lds_addr | global_addr[56:0] | type=2
    u32x4 g0;
    g0[0] = 1u;                                   // count=1, user mode
    g0[1] = lds_addr;                             // LDS byte address
    g0[2] = (unsigned int)ga;                     // global_addr[31:0]
    g0[3] = ((unsigned int)(ga >> 32) & 0x01FFFFFFu) | (2u << 30); // [56:32] | type=2
    // D# group 1
    unsigned int d0 = (1u << 16)      // data_size = 1 -> 2 bytes
                    | (1u << 20)      // pad_enable
                    | (3u << 22)      // pad_interval: 16 DWORDs (one 32-elem row)
                    | (7u << 25);     // pad_amount: 8 DWORDs (16 elems) -> stride 48
    unsigned int td0 = (unsigned int)row_stride_elems;   // tensor_dim0
    unsigned int td1 = 1u << 20;                         // tensor_dim1 (large)
    unsigned int d1 = (td0 & 0xFFFFu) << 16;             // tensor_dim0[15:0]
    unsigned int d2 = (td0 >> 16) | ((td1 & 0xFFFFu) << 16);
    unsigned int d3 = (td1 >> 16) | (32u << 16);         // tile_dim0 = 32
    unsigned int d4 = 128u;                              // tile_dim1 = 128, tile_dim2 = 0
    unsigned int d5 = (unsigned int)row_stride_elems;    // tensor_dim0_stride[31:0]
    i32x8 g1;
    g1[0] = (int)d0; g1[1] = (int)d1; g1[2] = (int)d2; g1[3] = (int)d3;
    g1[4] = (int)d4; g1[5] = (int)d5; g1[6] = 0; g1[7] = 0;
    i32x4 gz = {0, 0, 0, 0};
#if __clang_major__ >= 23
    i32x8 gz8 = {0, 0, 0, 0, 0, 0, 0, 0};
    __builtin_amdgcn_tensor_load_to_lds(g0, g1, gz, gz, gz8, 0);
#else
    __builtin_amdgcn_tensor_load_to_lds(g0, g1, gz, gz, 0);
#endif
}
#endif

__global__ __launch_bounds__(256) void gemm_bf16_nt(
    const __bf16* __restrict__ A, const __bf16* __restrict__ W,
    float* __restrict__ C, const float* __restrict__ addsrc,
    int M, int N, int K)
{
    __shared__ __bf16 As[2][128 * GPAD];
    __shared__ __bf16 Ws[2][128 * GPAD];
    const int m0 = blockIdx.y * 128;
    const int n0 = blockIdx.x * 128;
    const int tid = threadIdx.x, lane = tid & 31, wave = tid >> 5;
    const int wr = wave >> 2;     // 0..1 : 64-row strip
    const int wc = wave & 3;      // 0..3 : 32-col strip

    v8f acc[4][2];
#pragma unroll
    for (int i = 0; i < 4; ++i)
#pragma unroll
        for (int j = 0; j < 2; ++j) acc[i][j] = {};

    const int lrow = tid >> 1, lhalf = tid & 1;
    (void)lrow; (void)lhalf;

#if USE_TDM
    // Prologue: DMA the first A/W tile pair into buffer 0.
    if (wave == 0) {
        tdm_load_tile_128x32(A + (size_t)m0 * K,
                             (unsigned int)(uintptr_t)&As[0][0], K);
        tdm_load_tile_128x32(W + (size_t)n0 * K,
                             (unsigned int)(uintptr_t)&Ws[0][0], K);
    }
#endif

    for (int k0 = 0; k0 < K; k0 += 32) {
        const int ibuf = (k0 >> 5) & 1;
#if USE_TDM
        if (wave == 0) {
            if (k0 + 32 < K) {
                // Stream the next pair into the alternate buffer, then wait
                // only for the current pair (<=2 outstanding).
                tdm_load_tile_128x32(A + (size_t)m0 * K + k0 + 32,
                                     (unsigned int)(uintptr_t)&As[ibuf ^ 1][0], K);
                tdm_load_tile_128x32(W + (size_t)n0 * K + k0 + 32,
                                     (unsigned int)(uintptr_t)&Ws[ibuf ^ 1][0], K);
                __builtin_amdgcn_s_wait_tensorcnt((short)2);
            } else {
                __builtin_amdgcn_s_wait_tensorcnt((short)0);
            }
        }
        __syncthreads();
#else
        {
            const __bf16* srca = A + (size_t)(m0 + lrow) * K + k0 + lhalf * 16;
            *(v16bf*)&As[ibuf][lrow * GPAD + lhalf * 16] = *(const v16bf*)srca;
            int nr = n0 + lrow; if (nr > N - 1) nr = N - 1;
            const __bf16* srcw = W + (size_t)nr * K + k0 + lhalf * 16;
            *(v16bf*)&Ws[ibuf][lrow * GPAD + lhalf * 16] = *(const v16bf*)srcw;
            if (k0 + 32 < K) {
                __builtin_prefetch(srca + 32, 0, 1);
                __builtin_prefetch(srcw + 32, 0, 1);
            }
        }
        __syncthreads();
#endif
        v16bf bfrag[2];
#pragma unroll
        for (int nt = 0; nt < 2; ++nt) {
            int nrow = wc * 32 + nt * 16 + (lane & 15);
            bfrag[nt] = *(const v16bf*)&Ws[ibuf][nrow * GPAD + (lane >> 4) * 16];
        }
#pragma unroll
        for (int mt = 0; mt < 4; ++mt) {
            int mrow = wr * 64 + mt * 16 + (lane & 15);
            v16bf afrag = *(const v16bf*)&As[ibuf][mrow * GPAD + (lane >> 4) * 16];
#pragma unroll
            for (int nt = 0; nt < 2; ++nt)
                acc[mt][nt] = wmma_bf(afrag, bfrag[nt], acc[mt][nt]);
        }
        __syncthreads();
    }

#pragma unroll
    for (int mt = 0; mt < 4; ++mt)
#pragma unroll
        for (int nt = 0; nt < 2; ++nt)
#pragma unroll
            for (int r = 0; r < 8; ++r) {
                int m = m0 + wr * 64 + mt * 16 + (lane >> 4) * 8 + r;
                int n = n0 + wc * 32 + nt * 16 + (lane & 15);
                if (n < N) {
                    size_t idx = (size_t)m * N + n;
                    float v = acc[mt][nt][r];
                    if (addsrc) v += addsrc[idx];
                    C[idx] = v;
                }
            }
}

// ---------------------------------------------------------------------------
// Mamba prep: softplus(dt), a = -exp(A_log)*dt, x*dt (bf16), RMSNorm(B), RMSNorm(C)
// One block per token.
// ---------------------------------------------------------------------------
__global__ __launch_bounds__(256) void mamba_prep_k(
    const float* __restrict__ zx, const float* __restrict__ dt_bias,
    const float* __restrict__ A_log, const float* __restrict__ Bn_w,
    const float* __restrict__ Cn_w,
    __bf16* __restrict__ xdt, __bf16* __restrict__ Bb, __bf16* __restrict__ Cb,
    float* __restrict__ a_out)
{
    int t = blockIdx.x;
    const float* row = zx + (size_t)t * D_PROJ;
    __shared__ float dts[NHEADS];
    __shared__ float red[2];
    int tid = threadIdx.x;
    if (tid < 2) red[tid] = 0.f;
    __syncthreads();
    float bv = 0.f, cv = 0.f;
    if (tid < 128)      { bv = row[OFF_B + tid];        atomicAdd(&red[0], bv * bv); }
    else                { cv = row[OFF_C + (tid - 128)]; atomicAdd(&red[1], cv * cv); }
    __syncthreads();
    if (tid < 128) {
        float rstd = rsqrtf(red[0] * (1.f / 128.f) + 1e-5f);
        Bb[(size_t)t * D_STATE + tid] = (__bf16)(bv * rstd * Bn_w[tid]);
    } else {
        int i = tid - 128;
        float rstd = rsqrtf(red[1] * (1.f / 128.f) + 1e-5f);
        Cb[(size_t)t * D_STATE + i] = (__bf16)(cv * rstd * Cn_w[i]);
    }
    if (tid < NHEADS) {
        float x = row[OFF_DT + tid] + dt_bias[tid];
        float d = (x > 20.f) ? x : log1pf(__expf(x));
        dts[tid] = d;
        a_out[(size_t)t * NHEADS + tid] = -__expf(A_log[tid]) * d;
    }
    __syncthreads();
#pragma unroll
    for (int jj = 0; jj < 8; ++jj) {
        int idx = tid + jj * 256;
        float xv = row[OFF_XS + idx];
        xdt[(size_t)t * D_INNER + idx] = (__bf16)(xv * dts[idx >> 6]);
    }
}

// ---------------------------------------------------------------------------
// Per-chunk cumsum of a (decay logs): grid (c,h,b), 64 threads
// ---------------------------------------------------------------------------
__global__ __launch_bounds__(64) void chunk_cumsum_k(
    const float* __restrict__ a, float* __restrict__ Acum, float* __restrict__ Asum)
{
    int c = blockIdx.x, h = blockIdx.y, b = blockIdx.z;
    int l = threadIdx.x;
    __shared__ float ash[CHUNKL];
    int tok = b * SS_ + c * CHUNKL + l;
    ash[l] = a[(size_t)tok * NHEADS + h];
    __syncthreads();
    float s = 0.f;
    for (int j = 0; j <= l; ++j) s += ash[j];
    size_t base = (((size_t)b * NHEADS + h) * NCHUNK + c) * CHUNKL;
    Acum[base + l] = s;
    if (l == CHUNKL - 1) Asum[((size_t)b * NHEADS + h) * NCHUNK + c] = s;
}

// ---------------------------------------------------------------------------
// SSD per-chunk: Y_diag = (C Bᵀ ∘ L) Xdt ; states = Xdtᵀ(decay) B
// grid (c, h, b), 256 threads (8 waves), WMMA throughout.
// ---------------------------------------------------------------------------
__global__ __launch_bounds__(256) void ssd_chunk_k(
    const __bf16* __restrict__ Bb, const __bf16* __restrict__ Cb,
    const __bf16* __restrict__ xdt,
    const float* __restrict__ Acum, const float* __restrict__ Asum,
    float* __restrict__ ydiag, float* __restrict__ states)
{
    __shared__ __bf16 BTd[D_STATE * 80];  // [n][l], decay folded in
    __shared__ __bf16 XT [CHUNKL * 80];   // [p][l]
    __shared__ __bf16 Msh[CHUNKL * 80];   // [i][j] = CBᵀ ∘ L
    __shared__ float  Acsh[CHUNKL];
    __shared__ float  decsh[CHUNKL];

    int c = blockIdx.x, h = blockIdx.y, b = blockIdx.z;
    int tid = threadIdx.x, lane = tid & 31, wave = tid >> 5;
    int tok0 = b * SS_ + c * CHUNKL;
    size_t abase = (((size_t)b * NHEADS + h) * NCHUNK + c) * CHUNKL;

    if (tid < CHUNKL) {
        float ac = Acum[abase + tid];
        Acsh[tid] = ac;
        decsh[tid] = __expf(Asum[((size_t)b * NHEADS + h) * NCHUNK + c] - ac);
    }
    __syncthreads();

    for (int idx = tid; idx < CHUNKL * HEADDIM; idx += 256) {
        int l = idx >> 6, p = idx & 63;
        XT[p * 80 + l] = xdt[(size_t)(tok0 + l) * D_INNER + h * HEADDIM + p];
    }
    for (int idx = tid; idx < CHUNKL * D_STATE; idx += 256) {
        int l = idx >> 7, n = idx & 127;
        BTd[n * 80 + l] = (__bf16)((float)Bb[(size_t)(tok0 + l) * D_STATE + n] * decsh[l]);
    }
    __syncthreads();

    // CBᵀ (64x64), mask with L, store bf16 to Msh
    {
        int tt = wave * 2;
        for (int q = 0; q < 2; ++q, ++tt) {
            int ti = tt >> 2, tj = tt & 3;
            v8f acc = {};
#pragma unroll
            for (int kk = 0; kk < 4; ++kk) {
                v16bf af = *(const v16bf*)(Cb + (size_t)(tok0 + ti * 16 + (lane & 15)) * D_STATE
                                              + kk * 32 + (lane >> 4) * 16);
                v16bf bf_ = *(const v16bf*)(Bb + (size_t)(tok0 + tj * 16 + (lane & 15)) * D_STATE
                                               + kk * 32 + (lane >> 4) * 16);
                acc = wmma_bf(af, bf_, acc);
            }
#pragma unroll
            for (int r = 0; r < 8; ++r) {
                int i = ti * 16 + (lane >> 4) * 8 + r;
                int j = tj * 16 + (lane & 15);
                float Lv = (i >= j) ? __expf(Acsh[i] - Acsh[j]) : 0.f;
                Msh[i * 80 + j] = (__bf16)(acc[r] * Lv);
            }
        }
    }
    __syncthreads();

    // Y_diag = M @ X  (B-operand: XT rows = columns of X)
    {
        int tt = wave * 2;
        for (int q = 0; q < 2; ++q, ++tt) {
            int ti = tt >> 2, tp = tt & 3;
            v8f acc = {};
#pragma unroll
            for (int kk = 0; kk < 2; ++kk) {
                v16bf af  = *(const v16bf*)&Msh[(ti * 16 + (lane & 15)) * 80 + kk * 32 + (lane >> 4) * 16];
                v16bf bf_ = *(const v16bf*)&XT [(tp * 16 + (lane & 15)) * 80 + kk * 32 + (lane >> 4) * 16];
                acc = wmma_bf(af, bf_, acc);
            }
#pragma unroll
            for (int r = 0; r < 8; ++r) {
                int i = ti * 16 + (lane >> 4) * 8 + r;
                int p = tp * 16 + (lane & 15);
                ydiag[(size_t)(tok0 + i) * D_INNER + h * HEADDIM + p] = acc[r];
            }
        }
    }

    // states[p,n] = Σ_l X[l,p]·decay[l]·B[l,n]  (A: XT rows; B: BTd rows)
    {
        int tt = wave * 4;
        size_t sbase = (((size_t)b * NCHUNK + c) * NHEADS + h) * (HEADDIM * D_STATE);
        for (int q = 0; q < 4; ++q, ++tt) {
            int tp = tt >> 3, tn = tt & 7;
            v8f acc = {};
#pragma unroll
            for (int kk = 0; kk < 2; ++kk) {
                v16bf af  = *(const v16bf*)&XT [(tp * 16 + (lane & 15)) * 80 + kk * 32 + (lane >> 4) * 16];
                v16bf bf_ = *(const v16bf*)&BTd[(tn * 16 + (lane & 15)) * 80 + kk * 32 + (lane >> 4) * 16];
                acc = wmma_bf(af, bf_, acc);
            }
#pragma unroll
            for (int r = 0; r < 8; ++r) {
                int p = tp * 16 + (lane >> 4) * 8 + r;
                int n = tn * 16 + (lane & 15);
                states[sbase + (size_t)p * D_STATE + n] = acc[r];
            }
        }
    }
}

// ---------------------------------------------------------------------------
// Inter-chunk state scan: new_states[c] = state at start of chunk c.
// grid 64 = (b,h); 128 threads each own 64 contiguous (p,n) entries.
// ---------------------------------------------------------------------------
__global__ __launch_bounds__(128) void state_scan_k(
    const float* __restrict__ states, float* __restrict__ nstates,
    const float* __restrict__ Asum)
{
    int bh = blockIdx.x; int b = bh >> 5; int h = bh & 31;
    int tid = threadIdx.x;
    float acc[64];
#pragma unroll
    for (int i = 0; i < 64; ++i) acc[i] = 0.f;
    for (int c = 0; c < NCHUNK; ++c) {
        size_t base = (((size_t)b * NCHUNK + c) * NHEADS + h) * (HEADDIM * D_STATE)
                    + (size_t)tid * 64;
        float dec = __expf(Asum[((size_t)b * NHEADS + h) * NCHUNK + c]);
#pragma unroll
        for (int i = 0; i < 64; ++i) {
            nstates[base + i] = acc[i];
            acc[i] = acc[i] * dec + states[base + i];
        }
    }
}

// ---------------------------------------------------------------------------
// Y_off = C @ NSᵀ scaled by exp(Acum); fused + Y_diag + D-skip + SiLU(z) gate.
// grid (c, h, b), 256 threads.
// ---------------------------------------------------------------------------
__global__ __launch_bounds__(256) void ssd_yoff_k(
    const __bf16* __restrict__ Cb, const float* __restrict__ nstates,
    const float* __restrict__ Acum, const float* __restrict__ ydiag,
    const float* __restrict__ zx, const float* __restrict__ Dp,
    __bf16* __restrict__ ybf)
{
    __shared__ __bf16 NSsh[HEADDIM * 144];   // [p][n] bf16
    __shared__ float  Acsh[CHUNKL];
    int c = blockIdx.x, h = blockIdx.y, b = blockIdx.z;
    int tid = threadIdx.x, lane = tid & 31, wave = tid >> 5;
    int tok0 = b * SS_ + c * CHUNKL;
    size_t abase = (((size_t)b * NHEADS + h) * NCHUNK + c) * CHUNKL;
    if (tid < CHUNKL) Acsh[tid] = Acum[abase + tid];
    size_t nbase = (((size_t)b * NCHUNK + c) * NHEADS + h) * (HEADDIM * D_STATE);
    for (int idx = tid; idx < HEADDIM * D_STATE; idx += 256) {
        int p = idx >> 7, n = idx & 127;
        NSsh[p * 144 + n] = (__bf16)nstates[nbase + idx];
    }
    __syncthreads();
    float Dh = Dp[h];
    int tt = wave * 2;
    for (int q = 0; q < 2; ++q, ++tt) {
        int ti = tt >> 2, tp = tt & 3;
        v8f acc = {};
#pragma unroll
        for (int kk = 0; kk < 4; ++kk) {
            v16bf af  = *(const v16bf*)(Cb + (size_t)(tok0 + ti * 16 + (lane & 15)) * D_STATE
                                           + kk * 32 + (lane >> 4) * 16);
            v16bf bf_ = *(const v16bf*)&NSsh[(tp * 16 + (lane & 15)) * 144 + kk * 32 + (lane >> 4) * 16];
            acc = wmma_bf(af, bf_, acc);
        }
#pragma unroll
        for (int r = 0; r < 8; ++r) {
            int l = ti * 16 + (lane >> 4) * 8 + r;
            int p = tp * 16 + (lane & 15);
            int tok = tok0 + l;
            float yv = ydiag[(size_t)tok * D_INNER + h * HEADDIM + p]
                     + __expf(Acsh[l]) * acc[r];
            const float* row = zx + (size_t)tok * D_PROJ;
            yv += row[OFF_XS + h * HEADDIM + p] * Dh;
            float z = row[OFF_Z + h * HEADDIM + p];
            yv *= z / (1.f + __expf(-z));
            ybf[(size_t)tok * D_INNER + h * HEADDIM + p] = (__bf16)yv;
        }
    }
}

// ---------------------------------------------------------------------------
// QK RMSNorm + RoPE(16) + q_gain + 1/sqrt(64) fold, V passthrough, all -> bf16.
// One block per token; 16-lane groups per head; shfl_xor for norm + rope pair.
// ---------------------------------------------------------------------------
__global__ __launch_bounds__(256) void qkv_prep_k(
    const float* __restrict__ qf, const float* __restrict__ kf,
    const float* __restrict__ vf, const float* __restrict__ q_gain,
    __bf16* __restrict__ qb, __bf16* __restrict__ kb, __bf16* __restrict__ vb)
{
    int t = blockIdx.x;
    int pos = t & (SS_ - 1);
    int tid = threadIdx.x;
    int grp = tid >> 4;   // head group 0..15
    int li  = tid & 15;

    // Q heads
    {
        const float* src = qf + (size_t)t * (AH_ * AHD_) + grp * AHD_;
        float v0 = src[li], v1 = src[li + 16], v2 = src[li + 32], v3 = src[li + 48];
        float ss = v0 * v0 + v1 * v1 + v2 * v2 + v3 * v3;
        ss += __shfl_xor(ss, 1, 32); ss += __shfl_xor(ss, 2, 32);
        ss += __shfl_xor(ss, 4, 32); ss += __shfl_xor(ss, 8, 32);
        float rstd = rsqrtf(ss * (1.f / 64.f) + 1.1920929e-7f);
        v0 *= rstd; v1 *= rstd; v2 *= rstd; v3 *= rstd;
        float other = __shfl_xor(v0, 8, 32);
        int j = li & 7;
        float ang = (float)pos * __powf(10000.f, -(float)j * 0.125f);
        float s, cc; __sincosf(ang, &s, &cc);
        float r0 = (li < 8) ? (v0 * cc + other * s) : (-other * s + v0 * cc);
        float g = q_gain[grp] * 0.125f;
        __bf16* dst = qb + (size_t)t * (AH_ * AHD_) + grp * AHD_;
        dst[li] = (__bf16)(r0 * g); dst[li + 16] = (__bf16)(v1 * g);
        dst[li + 32] = (__bf16)(v2 * g); dst[li + 48] = (__bf16)(v3 * g);
    }
    // K heads (first 4 groups)
    if (grp < AKV_) {
        const float* src = kf + (size_t)t * (AKV_ * AHD_) + grp * AHD_;
        float v0 = src[li], v1 = src[li + 16], v2 = src[li + 32], v3 = src[li + 48];
        float ss = v0 * v0 + v1 * v1 + v2 * v2 + v3 * v3;
        ss += __shfl_xor(ss, 1, 32); ss += __shfl_xor(ss, 2, 32);
        ss += __shfl_xor(ss, 4, 32); ss += __shfl_xor(ss, 8, 32);
        float rstd = rsqrtf(ss * (1.f / 64.f) + 1.1920929e-7f);
        v0 *= rstd; v1 *= rstd; v2 *= rstd; v3 *= rstd;
        float other = __shfl_xor(v0, 8, 32);
        int j = li & 7;
        float ang = (float)pos * __powf(10000.f, -(float)j * 0.125f);
        float s, cc; __sincosf(ang, &s, &cc);
        float r0 = (li < 8) ? (v0 * cc + other * s) : (-other * s + v0 * cc);
        __bf16* dst = kb + (size_t)t * (AKV_ * AHD_) + grp * AHD_;
        dst[li] = (__bf16)r0; dst[li + 16] = (__bf16)v1;
        dst[li + 32] = (__bf16)v2; dst[li + 48] = (__bf16)v3;
    }
    // V passthrough
    vb[(size_t)t * (AKV_ * AHD_) + tid] = (__bf16)vf[(size_t)t * (AKV_ * AHD_) + tid];
}

// ---------------------------------------------------------------------------
// Flash attention (causal, GQA 16q/4kv, HD=64). grid (qtile, head, batch),
// 128 threads = 4 waves, each wave owns a 16-row query strip.
// QKᵀ and PV both on WMMA; online softmax via LDS row ownership.
// ---------------------------------------------------------------------------
__global__ __launch_bounds__(128) void attn_flash_k(
    const __bf16* __restrict__ Qb, const __bf16* __restrict__ Kb,
    const __bf16* __restrict__ Vb, __bf16* __restrict__ Ob)
{
    __shared__ __bf16 Ksh[64 * 80];
    __shared__ __bf16 VTsh[64 * 80];
    __shared__ float  Ssh[4][16 * 68];
    __shared__ __bf16 Psh[4][16 * 80];
    __shared__ float  alpha_sh[4][16];
    __shared__ float  linv_sh[4][16];

    const int b = blockIdx.z, h = blockIdx.y, qt = blockIdx.x;
    const int tid = threadIdx.x, lane = tid & 31, wave = tid >> 5;
    const int kvh = h >> 2;
    const int qrow0 = qt * 64 + wave * 16;

    v16bf qfr[2];
    {
        int tok = b * SS_ + qrow0 + (lane & 15);
        const __bf16* qp = Qb + (size_t)tok * (AH_ * AHD_) + h * AHD_ + (lane >> 4) * 16;
        qfr[0] = *(const v16bf*)qp;
        qfr[1] = *(const v16bf*)(qp + 32);
    }
    v8f oacc[4];
#pragma unroll
    for (int i = 0; i < 4; ++i) oacc[i] = {};
    float m_r = -3.0e30f, l_r = 0.f;   // valid in lanes 0..15

    for (int kt = 0; kt <= qt; ++kt) {
        {
            int r = tid >> 1, half = tid & 1;
            int tokk = b * SS_ + kt * 64 + r;
            const __bf16* kp = Kb + (size_t)tokk * (AKV_ * AHD_) + kvh * AHD_ + half * 32;
            *(v16bf*)&Ksh[r * 80 + half * 32]      = *(const v16bf*)kp;
            *(v16bf*)&Ksh[r * 80 + half * 32 + 16] = *(const v16bf*)(kp + 16);
            int d = r;
            const __bf16* vp = Vb + (size_t)(b * SS_ + kt * 64 + half * 32) * (AKV_ * AHD_)
                                  + kvh * AHD_ + d;
#pragma unroll
            for (int j = 0; j < 32; ++j)
                VTsh[d * 80 + half * 32 + j] = vp[(size_t)j * (AKV_ * AHD_)];
        }
        __syncthreads();

#pragma unroll
        for (int ct = 0; ct < 4; ++ct) {
            v8f sacc = {};
#pragma unroll
            for (int f = 0; f < 2; ++f) {
                v16bf bf_ = *(const v16bf*)&Ksh[(ct * 16 + (lane & 15)) * 80
                                                 + f * 32 + (lane >> 4) * 16];
                sacc = wmma_bf(qfr[f], bf_, sacc);
            }
#pragma unroll
            for (int r = 0; r < 8; ++r) {
                int row = (lane >> 4) * 8 + r;
                int col = ct * 16 + (lane & 15);
                float v = sacc[r];
                if (kt == qt) {
                    int qpos = qrow0 + row, kpos = kt * 64 + col;
                    if (kpos > qpos) v = -3.0e30f;
                }
                Ssh[wave][row * 68 + col] = v;
            }
        }
        __syncthreads();

        if (lane < 16) {
            const float* srow = &Ssh[wave][lane * 68];
            float mt = -3.0e30f;
            for (int j = 0; j < 64; ++j) mt = fmaxf(mt, srow[j]);
            float mnew  = fmaxf(m_r, mt);
            float alpha = __expf(m_r - mnew);
            float rs = 0.f;
            __bf16* prow = &Psh[wave][lane * 80];
            for (int j = 0; j < 64; ++j) {
                float pv = __expf(srow[j] - mnew);
                rs += pv;
                prow[j] = (__bf16)pv;
            }
            l_r = l_r * alpha + rs;
            m_r = mnew;
            alpha_sh[wave][lane] = alpha;
        }
        __syncthreads();

#pragma unroll
        for (int dt = 0; dt < 4; ++dt)
#pragma unroll
            for (int r = 0; r < 8; ++r)
                oacc[dt][r] *= alpha_sh[wave][(lane >> 4) * 8 + r];
#pragma unroll
        for (int f = 0; f < 2; ++f) {
            v16bf pf = *(const v16bf*)&Psh[wave][(lane & 15) * 80 + f * 32 + (lane >> 4) * 16];
#pragma unroll
            for (int dt = 0; dt < 4; ++dt) {
                v16bf vfr = *(const v16bf*)&VTsh[(dt * 16 + (lane & 15)) * 80
                                                  + f * 32 + (lane >> 4) * 16];
                oacc[dt] = wmma_bf(pf, vfr, oacc[dt]);
            }
        }
        __syncthreads();
    }

    if (lane < 16) linv_sh[wave][lane] = 1.f / l_r;
    __syncthreads();
#pragma unroll
    for (int dt = 0; dt < 4; ++dt)
#pragma unroll
        for (int r = 0; r < 8; ++r) {
            int row = (lane >> 4) * 8 + r;
            int tok = b * SS_ + qrow0 + row;
            int col = dt * 16 + (lane & 15);
            Ob[(size_t)tok * (AH_ * AHD_) + h * AHD_ + col] =
                (__bf16)(oacc[dt][r] * linv_sh[wave][row]);
        }
}

// ---------------------------------------------------------------------------
// Host-side pipeline
// ---------------------------------------------------------------------------
extern "C" void kernel_launch(void* const* d_in, const int* in_sizes, int n_in,
                              void* d_out, int out_size, void* d_ws, size_t ws_size,
                              hipStream_t stream)
{
    (void)in_sizes; (void)n_in; (void)out_size; (void)ws_size;
    const float* x        = (const float*)d_in[0];
    const float* mnorm_w  = (const float*)d_in[1];
    const float* in_w     = (const float*)d_in[2];
    const float* out_w    = (const float*)d_in[3];
    const float* D_param  = (const float*)d_in[4];
    const float* dt_bias  = (const float*)d_in[5];
    const float* A_log    = (const float*)d_in[6];
    const float* Bn_w     = (const float*)d_in[7];
    const float* Cn_w     = (const float*)d_in[8];
    const float* ln1_w    = (const float*)d_in[9];
    const float* ln1_b    = (const float*)d_in[10];
    const float* cq_w     = (const float*)d_in[11];
    const float* ck_w     = (const float*)d_in[12];
    const float* cv_w     = (const float*)d_in[13];
    const float* cproj_w  = (const float*)d_in[14];
    const float* q_gain   = (const float*)d_in[15];
    const float* ln2_w    = (const float*)d_in[16];
    const float* ln2_b    = (const float*)d_in[17];
    const float* fc_w     = (const float*)d_in[18];
    const float* proj_w   = (const float*)d_in[19];

    char* p = (char*)d_ws;
    auto alloc = [&](size_t bytes) -> void* {
        void* r = (void*)p;
        p += (bytes + 255) & ~(size_t)255;
        return r;
    };

    const size_t M = TOKENS;
    __bf16* wInB  = (__bf16*)alloc((size_t)D_PROJ * DD_ * 2);
    __bf16* wOutB = (__bf16*)alloc((size_t)DD_ * D_INNER * 2);
    __bf16* wQB   = (__bf16*)alloc((size_t)DD_ * DD_ * 2);
    __bf16* wKB   = (__bf16*)alloc((size_t)(AKV_*AHD_) * DD_ * 2);
    __bf16* wVB   = (__bf16*)alloc((size_t)(AKV_*AHD_) * DD_ * 2);
    __bf16* wPrB  = (__bf16*)alloc((size_t)DD_ * DD_ * 2);
    __bf16* wFcB  = (__bf16*)alloc((size_t)MLPD * DD_ * 2);
    __bf16* wPjB  = (__bf16*)alloc((size_t)DD_ * MLPD * 2);
    __bf16* hB    = (__bf16*)alloc(M * DD_ * 2);
    float*  zxb   = (float*) alloc(M * D_PROJ * 4);
    __bf16* xdtB  = (__bf16*)alloc(M * D_INNER * 2);
    __bf16* BBf   = (__bf16*)alloc(M * D_STATE * 2);
    __bf16* CBf   = (__bf16*)alloc(M * D_STATE * 2);
    float*  aF    = (float*) alloc(M * NHEADS * 4);
    float*  Acum  = (float*) alloc((size_t)BB_ * NHEADS * NCHUNK * CHUNKL * 4);
    float*  Asum  = (float*) alloc((size_t)BB_ * NHEADS * NCHUNK * 4);
    float*  states  = (float*)alloc((size_t)BB_ * NCHUNK * NHEADS * HEADDIM * D_STATE * 4);
    float*  nstates = (float*)alloc((size_t)BB_ * NCHUNK * NHEADS * HEADDIM * D_STATE * 4);
    float*  ydiag = (float*) alloc(M * D_INNER * 4);
    __bf16* ybf   = (__bf16*)alloc(M * D_INNER * 2);
    float*  x1    = (float*) alloc(M * DD_ * 4);

    // Aliased after last use of their hosts:
    float*  qf = zxb;                                   // after ssd_yoff, zxb free
    float*  kf = qf + M * DD_;
    float*  vf = kf + M * (AKV_ * AHD_);
    __bf16* qb2 = xdtB;                                 // after ssd_chunk
    __bf16* kb2 = qb2 + M * DD_;
    __bf16* vb2 = kb2 + M * (AKV_ * AHD_);
    __bf16* attnB = ybf;                                // after out_proj GEMM
    float*  x2 = ydiag;                                 // after ssd_yoff
    float*  fcF = states;                               // after state_scan
    __bf16* fcB = (__bf16*)nstates;                     // after ssd_yoff
    float*  outF = (float*)d_out;

    dim3 blk256(256), blk128(128), blk64(64);
    auto gcvt = [](size_t n) { return dim3((unsigned)((n + 255) / 256)); };

    // --- weight conversions (fp32 -> bf16) ---
    cvt_bf16_k<<<gcvt((size_t)D_PROJ*DD_), blk256, 0, stream>>>(in_w,   wInB,  (size_t)D_PROJ*DD_);
    cvt_bf16_k<<<gcvt((size_t)DD_*D_INNER), blk256, 0, stream>>>(out_w, wOutB, (size_t)DD_*D_INNER);
    cvt_bf16_k<<<gcvt((size_t)DD_*DD_), blk256, 0, stream>>>(cq_w,      wQB,   (size_t)DD_*DD_);
    cvt_bf16_k<<<gcvt((size_t)(AKV_*AHD_)*DD_), blk256, 0, stream>>>(ck_w, wKB, (size_t)(AKV_*AHD_)*DD_);
    cvt_bf16_k<<<gcvt((size_t)(AKV_*AHD_)*DD_), blk256, 0, stream>>>(cv_w, wVB, (size_t)(AKV_*AHD_)*DD_);
    cvt_bf16_k<<<gcvt((size_t)DD_*DD_), blk256, 0, stream>>>(cproj_w,   wPrB,  (size_t)DD_*DD_);
    cvt_bf16_k<<<gcvt((size_t)MLPD*DD_), blk256, 0, stream>>>(fc_w,     wFcB,  (size_t)MLPD*DD_);
    cvt_bf16_k<<<gcvt((size_t)DD_*MLPD), blk256, 0, stream>>>(proj_w,   wPjB,  (size_t)DD_*MLPD);

    // === Mamba block ===
    rmsnorm_bf16_k<<<dim3(TOKENS), blk256, 0, stream>>>(x, mnorm_w, hB, DD_);
    gemm_bf16_nt<<<dim3((D_PROJ + 127) / 128, TOKENS / 128), blk256, 0, stream>>>(
        hB, wInB, zxb, nullptr, TOKENS, D_PROJ, DD_);
    mamba_prep_k<<<dim3(TOKENS), blk256, 0, stream>>>(
        zxb, dt_bias, A_log, Bn_w, Cn_w, xdtB, BBf, CBf, aF);
    chunk_cumsum_k<<<dim3(NCHUNK, NHEADS, BB_), blk64, 0, stream>>>(aF, Acum, Asum);
    ssd_chunk_k<<<dim3(NCHUNK, NHEADS, BB_), blk256, 0, stream>>>(
        BBf, CBf, xdtB, Acum, Asum, ydiag, states);
    state_scan_k<<<dim3(BB_ * NHEADS), blk128, 0, stream>>>(states, nstates, Asum);
    ssd_yoff_k<<<dim3(NCHUNK, NHEADS, BB_), blk256, 0, stream>>>(
        CBf, nstates, Acum, ydiag, zxb, D_param, ybf);
    gemm_bf16_nt<<<dim3(DD_ / 128, TOKENS / 128), blk256, 0, stream>>>(
        ybf, wOutB, x1, x, TOKENS, DD_, D_INNER);

    // === Attention block ===
    layernorm_bf16_k<<<dim3(TOKENS), blk256, 0, stream>>>(x1, ln1_w, ln1_b, hB, DD_);
    gemm_bf16_nt<<<dim3(DD_ / 128, TOKENS / 128), blk256, 0, stream>>>(
        hB, wQB, qf, nullptr, TOKENS, DD_, DD_);
    gemm_bf16_nt<<<dim3((AKV_*AHD_ + 127) / 128, TOKENS / 128), blk256, 0, stream>>>(
        hB, wKB, kf, nullptr, TOKENS, AKV_*AHD_, DD_);
    gemm_bf16_nt<<<dim3((AKV_*AHD_ + 127) / 128, TOKENS / 128), blk256, 0, stream>>>(
        hB, wVB, vf, nullptr, TOKENS, AKV_*AHD_, DD_);
    qkv_prep_k<<<dim3(TOKENS), blk256, 0, stream>>>(qf, kf, vf, q_gain, qb2, kb2, vb2);
    attn_flash_k<<<dim3(SS_ / 64, AH_, BB_), blk128, 0, stream>>>(qb2, kb2, vb2, attnB);
    gemm_bf16_nt<<<dim3(DD_ / 128, TOKENS / 128), blk256, 0, stream>>>(
        attnB, wPrB, x2, x1, TOKENS, DD_, DD_);

    // === MLP ===
    layernorm_bf16_k<<<dim3(TOKENS), blk256, 0, stream>>>(x2, ln2_w, ln2_b, hB, DD_);
    gemm_bf16_nt<<<dim3(MLPD / 128, TOKENS / 128), blk256, 0, stream>>>(
        hB, wFcB, fcF, nullptr, TOKENS, MLPD, DD_);
    silu_cvt_k<<<gcvt(M * MLPD), blk256, 0, stream>>>(fcF, fcB, M * MLPD);
    gemm_bf16_nt<<<dim3(DD_ / 128, TOKENS / 128), blk256, 0, stream>>>(
        fcB, wPjB, outF, x2, TOKENS, DD_, MLPD);
}
